// GAT_79388175499823
// MI455X (gfx1250) — matrix operand order repaired and checked by reading, not compiled, not run on hardware
//
#include <hip/hip_runtime.h>
#include <math.h>

#define N_NODES  50000
#define N_EDGES  800000
#define IN_DIM   512
#define HID      64
#define H1_      4
#define OUT_DIM  100
#define D1       256      /* H1*HID */
#define N2PAD    112      /* OUT_DIM padded to multiple of 16 */
#define NEG_SLOPE 0.2f

typedef __bf16 bf16_t;
typedef bf16_t bf16x16 __attribute__((ext_vector_type(16)));
typedef float  floatx8 __attribute__((ext_vector_type(8)));

// ---------- helpers ----------
static __device__ __forceinline__ unsigned fenc(float f) {
  // order-preserving float -> uint map (for atomicMax segment-max)
  unsigned u = __float_as_uint(f);
  return (u & 0x80000000u) ? ~u : (u | 0x80000000u);
}
static __device__ __forceinline__ float fdec(unsigned u) {
  unsigned b = (u & 0x80000000u) ? (u & 0x7fffffffu) : ~u;
  return __uint_as_float(b);
}
static __device__ __forceinline__ float lrelu(float v) {
  return v > 0.0f ? v : NEG_SLOPE * v;
}

// ---------- generic fill ----------
__global__ void fill_u32_k(unsigned* __restrict__ p, unsigned v, size_t n) {
  size_t i = (size_t)blockIdx.x * blockDim.x + threadIdx.x;
  if (i < n) p[i] = v;
}

// ---------- pack weights: W[K,N] row-major -> Wt[Npad,K] bf16 (transposed) ----------
__global__ void pack_wt_k(const float* __restrict__ W, bf16_t* __restrict__ Wt,
                          int K, int N, int Npad) {
  int t = blockIdx.x * blockDim.x + threadIdx.x;
  if (t >= Npad * K) return;
  int n = t / K, k = t - n * K;
  float v = (n < N) ? W[(size_t)k * N + n] : 0.0f;
  Wt[(size_t)n * K + k] = (bf16_t)v;
}

// ---------- WMMA GEMM: C[M,ldc] = A[M,K](f32) * Bt[.,K](bf16, N-major) ----------
// block = 128 threads = 4 waves; wave w -> rows 16*(4*by+w); block covers NT*16 cols
// starting at nbase0 + bx*NT*16. Software-pipelined: B loads -> A convert ->
// A prefetch (next kb) -> NT WMMAs.
template <int NT>
__global__ void gemm_wmma_bf16(const float* __restrict__ A,
                               const bf16_t* __restrict__ Bt,
                               float* __restrict__ C,
                               int M, int K, int nbase0, int ldc) {
  const int wave = (int)(threadIdx.x >> 5);
  const int lane = (int)(threadIdx.x & 31);
  const int mbase = (blockIdx.y * 4 + wave) * 16;
  if (mbase >= M) return;                         // wave-uniform guard
  const int nbase = nbase0 + (int)blockIdx.x * (NT * 16);

  const int r  = lane & 15;
  const int hi = lane >> 4;                       // half-wave select

  floatx8 acc[NT] = {};

  // A: lane<16 holds K {0..15}; lane>=16 holds K {16..31}
  const float* arow = A + (size_t)(mbase + r) * K + hi * 16;
  // B: lane holds column nbase+t*16+(lane&15); 16 contiguous K (offset 16 for hi half)
  const bf16_t* brow = Bt + (size_t)(nbase + r) * K + hi * 16;

  // prologue: first A chunk in flight
  float4 f0 = *(const float4*)(arow + 0);
  float4 f1 = *(const float4*)(arow + 4);
  float4 f2 = *(const float4*)(arow + 8);
  float4 f3 = *(const float4*)(arow + 12);

  for (int kb = 0; kb < K; kb += 32) {
    // issue all B fragment loads for this K-step
    bf16x16 bv[NT];
    #pragma unroll
    for (int t = 0; t < NT; ++t)
      bv[t] = *(const bf16x16*)(brow + (size_t)t * 16 * K + kb);

    // convert the A chunk loaded in the previous iteration
    bf16x16 av;
    av[0]=(bf16_t)f0.x;  av[1]=(bf16_t)f0.y;  av[2]=(bf16_t)f0.z;  av[3]=(bf16_t)f0.w;
    av[4]=(bf16_t)f1.x;  av[5]=(bf16_t)f1.y;  av[6]=(bf16_t)f1.z;  av[7]=(bf16_t)f1.w;
    av[8]=(bf16_t)f2.x;  av[9]=(bf16_t)f2.y;  av[10]=(bf16_t)f2.z; av[11]=(bf16_t)f2.w;
    av[12]=(bf16_t)f3.x; av[13]=(bf16_t)f3.y; av[14]=(bf16_t)f3.z; av[15]=(bf16_t)f3.w;

    // prefetch next K-step's A chunk (hidden behind the WMMA group)
    if (kb + 32 < K) {
      f0 = *(const float4*)(arow + kb + 32);
      f1 = *(const float4*)(arow + kb + 36);
      f2 = *(const float4*)(arow + kb + 40);
      f3 = *(const float4*)(arow + kb + 44);
    }

    #pragma unroll
    for (int t = 0; t < NT; ++t)
      acc[t] = __builtin_amdgcn_wmma_f32_16x16x32_bf16(
          false, av, false, bv[t], (short)0, acc[t], false, false);
  }

  // C/D: VGPR i -> row mbase + i + 8*hi, col nbase + t*16 + (lane&15)
  #pragma unroll
  for (int t = 0; t < NT; ++t) {
    float* cp = C + (size_t)(mbase + hi * 8) * ldc + nbase + t * 16 + r;
    #pragma unroll
    for (int i = 0; i < 8; ++i) cp[(size_t)i * ldc] = acc[t][i];
  }
}

// ---------- per-node attention scores: el/er[n*H+h] = <feat[n,h,:], attn[h,:]> ----------
__global__ void node_scores_k(const float* __restrict__ feat,
                              const float* __restrict__ al, const float* __restrict__ ar,
                              float* __restrict__ el, float* __restrict__ er,
                              int Nn, int H, int F, int ldf) {
  int t = blockIdx.x * blockDim.x + threadIdx.x;
  if (t >= Nn * H) return;
  int n = t / H, h = t - n * H;
  const float* fp = feat + (size_t)n * ldf + h * F;
  const float* ap = al + h * F;
  const float* bp = ar + h * F;
  float sl = 0.f, sr = 0.f;
  for (int f = 0; f < F; ++f) {
    float v = fp[f];
    sl = fmaf(v, ap[f], sl);
    sr = fmaf(v, bp[f], sr);
  }
  el[t] = sl; er[t] = sr;
}

// ---------- edge pass 1: segment max (ordered-uint atomicMax) ----------
__global__ void edge_max_k(const int* __restrict__ src, const int* __restrict__ dst,
                           const float* __restrict__ el, const float* __restrict__ er,
                           unsigned* __restrict__ mkey, int E, int H) {
  int e = blockIdx.x * blockDim.x + threadIdx.x;
  if (e >= E) return;
  int s = src[e], d = dst[e];
  for (int h = 0; h < H; ++h) {
    float v = lrelu(el[s * H + h] + er[d * H + h]);
    atomicMax(&mkey[d * H + h], fenc(v));
  }
}

// ---------- edge pass 2: segment sum of exp(e - m) ----------
__global__ void edge_expsum_k(const int* __restrict__ src, const int* __restrict__ dst,
                              const float* __restrict__ el, const float* __restrict__ er,
                              const unsigned* __restrict__ mkey, float* __restrict__ ssum,
                              int E, int H) {
  int e = blockIdx.x * blockDim.x + threadIdx.x;
  if (e >= E) return;
  int s = src[e], d = dst[e];
  for (int h = 0; h < H; ++h) {
    float v = lrelu(el[s * H + h] + er[d * H + h]);
    float m = fdec(mkey[d * H + h]);
    atomicAdd(&ssum[d * H + h], __expf(v - m));
  }
}

// ---------- edge pass 3: out[dst,h,:] += a * feat[src,h,:] ----------
__global__ void edge_acc_k(const int* __restrict__ src, const int* __restrict__ dst,
                           const float* __restrict__ el, const float* __restrict__ er,
                           const unsigned* __restrict__ mkey, const float* __restrict__ ssum,
                           const float* __restrict__ feat, float* __restrict__ outp,
                           int E, int H, int F, int ldf, int ldo) {
  int t = blockIdx.x * blockDim.x + threadIdx.x;
  if (t >= E * H) return;
  int e = t / H, h = t - e * H;
  int s = src[e], d = dst[e];
  float v = lrelu(el[s * H + h] + er[d * H + h]);
  float m = fdec(mkey[d * H + h]);
  float a = __expf(v - m) / ssum[d * H + h];
  const float* fp = feat + (size_t)s * ldf + h * F;
  float* op = outp + (size_t)d * ldo + h * F;
  for (int f = 0; f < F; f += 4) {
    float4 x = *(const float4*)(fp + f);
    atomicAdd(op + f + 0, x.x * a);
    atomicAdd(op + f + 1, x.y * a);
    atomicAdd(op + f + 2, x.z * a);
    atomicAdd(op + f + 3, x.w * a);
  }
}

// ---------- x = relu(x + bias) in place ----------
__global__ void relu_bias_k(float* __restrict__ x, const float* __restrict__ b,
                            int ld, size_t n) {
  size_t i = (size_t)blockIdx.x * blockDim.x + threadIdx.x;
  if (i >= n) return;
  x[i] = fmaxf(x[i] + b[i % (size_t)ld], 0.0f);
}

// ---------- x += bias ----------
__global__ void bias_add_k(float* __restrict__ x, const float* __restrict__ b,
                           int ld, size_t n) {
  size_t i = (size_t)blockIdx.x * blockDim.x + threadIdx.x;
  if (i >= n) return;
  x[i] += b[i % (size_t)ld];
}

static inline unsigned cdiv(size_t a, unsigned b) { return (unsigned)((a + b - 1) / b); }

extern "C" void kernel_launch(void* const* d_in, const int* in_sizes, int n_in,
                              void* d_out, int out_size, void* d_ws, size_t ws_size,
                              hipStream_t stream) {
  const float* h   = (const float*)d_in[0];
  const int*   src = (const int*)d_in[1];
  const int*   dst = (const int*)d_in[2];
  const float* W1  = (const float*)d_in[3];
  const float* al1 = (const float*)d_in[4];
  const float* ar1 = (const float*)d_in[5];
  const float* b1  = (const float*)d_in[6];
  const float* W2  = (const float*)d_in[7];
  const float* al2 = (const float*)d_in[8];
  const float* ar2 = (const float*)d_in[9];
  const float* b2  = (const float*)d_in[10];
  float* out = (float*)d_out;

  char* ws = (char*)d_ws;
  size_t off = 0;
  auto alloc = [&](size_t bytes) -> void* {
    void* p = (void*)(ws + off);
    off += (bytes + 255) & ~(size_t)255;
    return p;
  };
  float*   feat1 = (float*)  alloc((size_t)N_NODES * D1 * 4);
  float*   out1  = (float*)  alloc((size_t)N_NODES * D1 * 4);   // becomes x after relu
  float*   feat2 = (float*)  alloc((size_t)N_NODES * N2PAD * 4);
  bf16_t*  W1t   = (bf16_t*) alloc((size_t)D1 * IN_DIM * 2);
  bf16_t*  W2t   = (bf16_t*) alloc((size_t)N2PAD * D1 * 2);
  float*   el1   = (float*)  alloc((size_t)N_NODES * H1_ * 4);
  float*   er1   = (float*)  alloc((size_t)N_NODES * H1_ * 4);
  unsigned* m1   = (unsigned*)alloc((size_t)N_NODES * H1_ * 4);
  float*   s1    = (float*)  alloc((size_t)N_NODES * H1_ * 4);
  float*   el2   = (float*)  alloc((size_t)N_NODES * 4);
  float*   er2   = (float*)  alloc((size_t)N_NODES * 4);
  unsigned* m2   = (unsigned*)alloc((size_t)N_NODES * 4);
  float*   s2    = (float*)  alloc((size_t)N_NODES * 4);
  (void)ws_size; (void)in_sizes; (void)n_in; (void)out_size;

  const unsigned TB = 256;
  const int E = N_EDGES;

  // ---- zero-init accumulators (every call: graph replay safe) ----
  fill_u32_k<<<cdiv((size_t)N_NODES * D1, TB), TB, 0, stream>>>((unsigned*)out1, 0u, (size_t)N_NODES * D1);
  fill_u32_k<<<cdiv((size_t)N_NODES * OUT_DIM, TB), TB, 0, stream>>>((unsigned*)out, 0u, (size_t)N_NODES * OUT_DIM);
  fill_u32_k<<<cdiv((size_t)N_NODES * H1_, TB), TB, 0, stream>>>(m1, 0u, (size_t)N_NODES * H1_);
  fill_u32_k<<<cdiv((size_t)N_NODES * H1_, TB), TB, 0, stream>>>((unsigned*)s1, 0u, (size_t)N_NODES * H1_);
  fill_u32_k<<<cdiv((size_t)N_NODES, TB), TB, 0, stream>>>(m2, 0u, (size_t)N_NODES);
  fill_u32_k<<<cdiv((size_t)N_NODES, TB), TB, 0, stream>>>((unsigned*)s2, 0u, (size_t)N_NODES);

  // ---- pack weights (transpose -> bf16, N-major) ----
  pack_wt_k<<<cdiv((size_t)D1 * IN_DIM, TB), TB, 0, stream>>>(W1, W1t, IN_DIM, D1, D1);
  pack_wt_k<<<cdiv((size_t)N2PAD * D1, TB), TB, 0, stream>>>(W2, W2t, D1, OUT_DIM, N2PAD);

  const unsigned mblocks = (N_NODES / 16 + 3) / 4;   // 3125 row tiles -> 782

  // ---- layer 1 ----
  gemm_wmma_bf16<4><<<dim3(D1 / 64, mblocks), 128, 0, stream>>>(h, W1t, feat1, N_NODES, IN_DIM, 0, D1);
  node_scores_k<<<cdiv((size_t)N_NODES * H1_, TB), TB, 0, stream>>>(feat1, al1, ar1, el1, er1, N_NODES, H1_, HID, D1);
  edge_max_k<<<cdiv((size_t)E, TB), TB, 0, stream>>>(src, dst, el1, er1, m1, E, H1_);
  edge_expsum_k<<<cdiv((size_t)E, TB), TB, 0, stream>>>(src, dst, el1, er1, m1, s1, E, H1_);
  edge_acc_k<<<cdiv((size_t)E * H1_, TB), TB, 0, stream>>>(src, dst, el1, er1, m1, s1, feat1, out1, E, H1_, HID, D1, D1);
  relu_bias_k<<<cdiv((size_t)N_NODES * D1, TB), TB, 0, stream>>>(out1, b1, D1, (size_t)N_NODES * D1);

  // ---- layer 2 (x = out1); N2PAD=112 split into branchless 64 + 48 col launches ----
  gemm_wmma_bf16<4><<<dim3(1, mblocks), 128, 0, stream>>>(out1, W2t, feat2, N_NODES, D1, 0,  N2PAD);
  gemm_wmma_bf16<3><<<dim3(1, mblocks), 128, 0, stream>>>(out1, W2t, feat2, N_NODES, D1, 64, N2PAD);
  node_scores_k<<<cdiv((size_t)N_NODES, TB), TB, 0, stream>>>(feat2, al2, ar2, el2, er2, N_NODES, 1, OUT_DIM, N2PAD);
  edge_max_k<<<cdiv((size_t)E, TB), TB, 0, stream>>>(src, dst, el2, er2, m2, E, 1);
  edge_expsum_k<<<cdiv((size_t)E, TB), TB, 0, stream>>>(src, dst, el2, er2, m2, s2, E, 1);
  edge_acc_k<<<cdiv((size_t)E, TB), TB, 0, stream>>>(src, dst, el2, er2, m2, s2, feat2, out, E, 1, OUT_DIM, N2PAD, OUT_DIM);
  bias_add_k<<<cdiv((size_t)N_NODES * OUT_DIM, TB), TB, 0, stream>>>(out, b2, OUT_DIM, (size_t)N_NODES * OUT_DIM);
}
